// TorchLMHeadGRPO_62801011802684
// MI455X (gfx1250) — matrix operand order repaired and checked by reading, not compiled, not run on hardware
//
#include <hip/hip_runtime.h>
#include <hip/hip_fp16.h>

// Problem constants (from reference)
#define Bb   4
#define Tt   512
#define Hh   1024
#define Vv   32000
#define BT   2048          // B*T
#define NCOLS 256          // V columns per workgroup (kernel 1)
#define NCHUNK 125         // Vv / NCOLS
#define MROWS 128          // tokens per workgroup (kernel 1)
#define BETA 0.1f

typedef __attribute__((ext_vector_type(16))) _Float16 v16h;
typedef __attribute__((ext_vector_type(8)))  float    v8f;

// ---------------------------------------------------------------------------
// Kernel 1: f16x3-split WMMA GEMM computing per-token partial (max, sumexp)
// over a 256-column chunk of V, for model = blockIdx.z
// (0 = policy (x, W), 1 = reference (ref_input, ref_W)).
//
// Workgroup: 256 threads (8 waves), tile = M128 x N256, K loop 1024 step 32.
// Double-buffered LDS fragment staging, ONE barrier per k-step: the whole
// block cooperatively converts the 256x32 W slab to f16 hi/lo fragment
// planes (one column per thread); next slab's global loads + conversion
// overlap the current WMMA stream. Each wave: 2 m-tiles x 8 n-tiles x 3
// WMMAs (hi*hi + hi*lo + lo*hi, ~22 mantissa bits ~= fp32).
// ---------------------------------------------------------------------------
__global__ __launch_bounds__(256)
void grpo_lse_partials(const float* __restrict__ x,
                       const float* __restrict__ ref_input,
                       const float* __restrict__ W,
                       const float* __restrict__ ref_W,
                       float* __restrict__ partials)
{
    const int chunk    = blockIdx.x;          // 0..124
    const int tokBlock = blockIdx.y;          // 0..15 (128 tokens each)
    const int model    = blockIdx.z;          // 0..1

    const float* __restrict__ Xp = model ? ref_input : x;
    const float* __restrict__ Wp = model ? ref_W     : W;

    const int tid   = threadIdx.x;
    const int lane  = tid & 31;
    const int wave  = tid >> 5;
    const int mbase = (wave & 3) * 2;         // first of this wave's 2 m-tiles
    const int ng    = wave >> 2;              // 0..1 (n-group of 128 cols)

    const int tok0 = tokBlock * MROWS;
    const int col0 = chunk * NCOLS;

    // Double-buffered B fragment staging: [buf][n-tile][lane][16 halfs]
    __shared__ _Float16 Bhi[2][16][32][16] __attribute__((aligned(32)));
    __shared__ _Float16 Blo[2][16][32][16] __attribute__((aligned(32)));

    // conversion role: thread t owns column col0 + t
    const float* __restrict__ wcol = Wp + (size_t)(col0 + tid) * Hh;
    const int wtile = tid >> 4;               // n-tile this thread fills
    const int wcl   = tid & 15;               // column-within-tile

    // compute role: A rows for this wave's two m-tiles
    const bool hiHalf = (lane >= 16);
    const int  row    = lane & 15;
    const float* __restrict__ arow[2] = {
        Xp + (size_t)(tok0 + (mbase + 0) * 16 + row) * Hh,
        Xp + (size_t)(tok0 + (mbase + 1) * 16 + row) * Hh
    };

    v8f acc[2][8];
    const v8f vzero = {0.f,0.f,0.f,0.f,0.f,0.f,0.f,0.f};
#pragma unroll
    for (int mt = 0; mt < 2; ++mt)
#pragma unroll
        for (int nt = 0; nt < 8; ++nt) acc[mt][nt] = vzero;

    // ---- prologue: load + convert slab kk=0
    v16h h0, h1, l0, l1;
    {
        float wf[32];
#pragma unroll
        for (int i = 0; i < 8; ++i) {
            float4 v = *(const float4*)(wcol + 4 * i);
            wf[4*i] = v.x; wf[4*i+1] = v.y; wf[4*i+2] = v.z; wf[4*i+3] = v.w;
        }
#pragma unroll
        for (int e = 0; e < 16; ++e) {
            _Float16 h = (_Float16)wf[e];
            h0[e] = h;  l0[e] = (_Float16)(wf[e] - (float)h);
            _Float16 g = (_Float16)wf[16 + e];
            h1[e] = g;  l1[e] = (_Float16)(wf[16 + e] - (float)g);
        }
    }

    for (int kk = 0; kk < Hh; kk += 32) {
        const int p = (kk >> 5) & 1;
        // B layout: lanes 0-15 hold K=kk..kk+15 (elem = K%16), lanes 16-31
        // hold K=kk+16..kk+31, column = lane%16.
        *(v16h*)&Bhi[p][wtile][wcl][0]      = h0;
        *(v16h*)&Bhi[p][wtile][wcl + 16][0] = h1;
        *(v16h*)&Blo[p][wtile][wcl][0]      = l0;
        *(v16h*)&Blo[p][wtile][wcl + 16][0] = l1;
        __syncthreads();   // single barrier per k-step (double buffering)

        // ---- issue next slab's global loads (in flight under the WMMAs)
        const bool more = (kk + 32) < Hh;
        float wf[32];
        if (more) {
#pragma unroll
            for (int i = 0; i < 8; ++i) {
                float4 v = *(const float4*)(wcol + kk + 32 + 4 * i);
                wf[4*i] = v.x; wf[4*i+1] = v.y; wf[4*i+2] = v.z; wf[4*i+3] = v.w;
            }
            if (kk + 64 < Hh)
                __builtin_prefetch(wcol + kk + 64, 0, 1);   // global_prefetch_b8
        }

        // ---- A fragments (lanes<16 hold K=kk+{0..15},
        // lanes>=16 hold K=kk+{16..31})
        v16h ah[2], al[2];
        const int kb = kk + (hiHalf ? 16 : 0);
#pragma unroll
        for (int mt = 0; mt < 2; ++mt) {
            const float* ar = arow[mt];
            float4 a0 = *(const float4*)(ar + kb);
            float4 a1 = *(const float4*)(ar + kb + 4);
            float4 a2 = *(const float4*)(ar + kb + 8);
            float4 a3 = *(const float4*)(ar + kb + 12);
            float af[16] = {a0.x,a0.y,a0.z,a0.w, a1.x,a1.y,a1.z,a1.w,
                            a2.x,a2.y,a2.z,a2.w, a3.x,a3.y,a3.z,a3.w};
#pragma unroll
            for (int e = 0; e < 16; ++e) {
                _Float16 h = (_Float16)af[e];
                ah[mt][e] = h;
                al[mt][e] = (_Float16)(af[e] - (float)h);
            }
        }

        // ---- WMMA: 8 n-tiles x 2 m-tiles x 3 (hi*hi + hi*lo + lo*hi)
#pragma unroll
        for (int nt = 0; nt < 8; ++nt) {
            v16h bh = *(const v16h*)&Bhi[p][ng * 8 + nt][lane][0];
            v16h bl = *(const v16h*)&Blo[p][ng * 8 + nt][lane][0];
#pragma unroll
            for (int mt = 0; mt < 2; ++mt) {
                acc[mt][nt] = __builtin_amdgcn_wmma_f32_16x16x32_f16(
                    false, ah[mt], false, bh, (short)0, acc[mt][nt], false, false);
                acc[mt][nt] = __builtin_amdgcn_wmma_f32_16x16x32_f16(
                    false, ah[mt], false, bl, (short)0, acc[mt][nt], false, false);
                acc[mt][nt] = __builtin_amdgcn_wmma_f32_16x16x32_f16(
                    false, al[mt], false, bh, (short)0, acc[mt][nt], false, false);
            }
        }

        // ---- convert next slab (overlaps the WMMA tail via scheduler)
        if (more) {
#pragma unroll
            for (int e = 0; e < 16; ++e) {
                _Float16 h = (_Float16)wf[e];
                h0[e] = h;  l0[e] = (_Float16)(wf[e] - (float)h);
                _Float16 g = (_Float16)wf[16 + e];
                h1[e] = g;  l1[e] = (_Float16)(wf[16 + e] - (float)g);
            }
        }
    }

    // ------------------------------------------------------------------
    // Per-row (per-token) max / sumexp over this chunk's 256 columns.
    // C layout: VGPR r holds row r (lanes 0-15) / row 8+r (lanes 16-31),
    // column = lane & 15.
    // ------------------------------------------------------------------
    __shared__ float redmax[2][8][16];
    __shared__ float redsum[2][8][16];

    float rmax[2][8];
#pragma unroll
    for (int mt = 0; mt < 2; ++mt)
#pragma unroll
        for (int r = 0; r < 8; ++r) {
            float v = acc[mt][0][r];
#pragma unroll
            for (int nt = 1; nt < 8; ++nt) v = fmaxf(v, acc[mt][nt][r]);
            for (int off = 1; off < 16; off <<= 1)
                v = fmaxf(v, __shfl_xor(v, off, 32));
            rmax[mt][r] = v;
        }
    if (lane == 0) {
#pragma unroll
        for (int mt = 0; mt < 2; ++mt)
#pragma unroll
            for (int r = 0; r < 8; ++r) redmax[ng][mbase + mt][r] = rmax[mt][r];
    }
    if (lane == 16) {
#pragma unroll
        for (int mt = 0; mt < 2; ++mt)
#pragma unroll
            for (int r = 0; r < 8; ++r) redmax[ng][mbase + mt][8 + r] = rmax[mt][r];
    }
    __syncthreads();

    const int rbase = hiHalf ? 8 : 0;
    float rsum[2][8];
#pragma unroll
    for (int mt = 0; mt < 2; ++mt)
#pragma unroll
        for (int r = 0; r < 8; ++r) {
            float cm = fmaxf(redmax[0][mbase + mt][rbase + r],
                             redmax[1][mbase + mt][rbase + r]);
            float s = 0.f;
#pragma unroll
            for (int nt = 0; nt < 8; ++nt) s += __expf(acc[mt][nt][r] - cm);
            for (int off = 1; off < 16; off <<= 1)
                s += __shfl_xor(s, off, 32);
            rsum[mt][r] = s;
        }
    if (lane == 0) {
#pragma unroll
        for (int mt = 0; mt < 2; ++mt)
#pragma unroll
            for (int r = 0; r < 8; ++r) redsum[ng][mbase + mt][r] = rsum[mt][r];
    }
    if (lane == 16) {
#pragma unroll
        for (int mt = 0; mt < 2; ++mt)
#pragma unroll
            for (int r = 0; r < 8; ++r) redsum[ng][mbase + mt][8 + r] = rsum[mt][r];
    }
    __syncthreads();

    // one wave per m-tile pair writes the per-token partials for this chunk
    if (ng == 0 && lane < 16) {
#pragma unroll
        for (int mt = 0; mt < 2; ++mt) {
            const int mtile = mbase + mt;
            const int token = tok0 + mtile * 16 + lane;
            float m = fmaxf(redmax[0][mtile][lane], redmax[1][mtile][lane]);
            float s = redsum[0][mtile][lane] + redsum[1][mtile][lane];
            size_t idx = (((size_t)model * BT + token) * NCHUNK + chunk) * 2;
            partials[idx]     = m;
            partials[idx + 1] = s;
        }
    }
}

// ---------------------------------------------------------------------------
// Kernel 2: one wave per (token, model). Selected-token logit via direct dot,
// then deterministic merge of the 125 (max, sumexp) partials -> logp.
// ---------------------------------------------------------------------------
__global__ __launch_bounds__(32)
void grpo_finalize_logp(const float* __restrict__ x,
                        const float* __restrict__ ref_input,
                        const float* __restrict__ W,
                        const float* __restrict__ ref_W,
                        const int*   __restrict__ ids,
                        const float* __restrict__ partials,
                        float* __restrict__ logps)
{
    const int token = blockIdx.x;   // 0..2047
    const int model = blockIdx.y;   // 0..1
    const int lane  = threadIdx.x;

    const float* __restrict__ Xp = model ? ref_input : x;
    const float* __restrict__ Wp = model ? ref_W     : W;

    // selected-token logit: dot(h, W[id]) over H=1024
    const int id = ids[token];
    const float* __restrict__ hv = Xp + (size_t)token * Hh;
    const float* __restrict__ wv = Wp + (size_t)id * Hh;
    float dot = 0.f;
    for (int k = lane; k < Hh; k += 32) dot += hv[k] * wv[k];
    for (int off = 1; off < 32; off <<= 1) dot += __shfl_xor(dot, off, 32);

    // merge (m, s) partials: lane-local sequential, then shfl tree
    const float* __restrict__ base =
        partials + (((size_t)model * BT + token) * NCHUNK) * 2;
    float m = -1e30f, s = 0.f;
    for (int c = lane; c < NCHUNK; c += 32) {
        float pm = base[2 * c], ps = base[2 * c + 1];
        float nm = fmaxf(m, pm);
        s = s * __expf(m - nm) + ps * __expf(pm - nm);
        m = nm;
    }
    for (int off = 1; off < 32; off <<= 1) {
        float om = __shfl_xor(m, off, 32);
        float os = __shfl_xor(s, off, 32);
        float nm = fmaxf(m, om);
        s = s * __expf(m - nm) + os * __expf(om - nm);
        m = nm;
    }

    if (lane == 0) {
        float lse = m + __logf(s);
        logps[(size_t)model * BT + token] = dot - lse;
    }
}

// ---------------------------------------------------------------------------
// Kernel 3: single block; deterministic tree reduction to the scalar loss.
// Forward of GRPO: coef_1 == 1 -> per-token loss = -adv + BETA*(exp(d)-d-1).
// ---------------------------------------------------------------------------
__global__ __launch_bounds__(256)
void grpo_loss_reduce(const float* __restrict__ logps,
                      const int*   __restrict__ attn_mask,
                      const float* __restrict__ advantages,
                      float* __restrict__ out)
{
    const int tid = threadIdx.x;
    float accL = 0.f, accM = 0.f;
    for (int t = tid; t < BT; t += 256) {
        float lp   = logps[t];
        float lr   = logps[BT + t];
        float msk  = (float)attn_mask[t];
        float adv  = advantages[t / Tt];
        float d    = lr - lp;
        float kl   = __expf(d) - d - 1.0f;
        float ptl  = -adv + BETA * kl;
        accL += ptl * msk;
        accM += msk;
    }
    __shared__ float sL[256], sM[256];
    sL[tid] = accL; sM[tid] = accM;
    __syncthreads();
    for (int s = 128; s > 0; s >>= 1) {
        if (tid < s) { sL[tid] += sL[tid + s]; sM[tid] += sM[tid + s]; }
        __syncthreads();
    }
    if (tid == 0) out[0] = sL[0] / fmaxf(sM[0], 1.0f);
}

// ---------------------------------------------------------------------------
extern "C" void kernel_launch(void* const* d_in, const int* in_sizes, int n_in,
                              void* d_out, int out_size, void* d_ws, size_t ws_size,
                              hipStream_t stream)
{
    // setup_inputs order: x, selected_token_ids, attention_mask, advantages,
    //                     ref_input, W, ref_W
    const float* x          = (const float*)d_in[0];
    const int*   ids        = (const int*)  d_in[1];
    const int*   attn       = (const int*)  d_in[2];
    const float* advantages = (const float*)d_in[3];
    const float* ref_input  = (const float*)d_in[4];
    const float* W          = (const float*)d_in[5];
    const float* ref_W      = (const float*)d_in[6];

    float* ws       = (float*)d_ws;
    float* partials = ws;                                   // 2*BT*NCHUNK*2 floats (4 MB)
    float* logps    = ws + (size_t)2 * BT * NCHUNK * 2;     // 2*BT floats

    dim3 g1(NCHUNK, BT / MROWS, 2);
    grpo_lse_partials<<<g1, 256, 0, stream>>>(x, ref_input, W, ref_W, partials);

    dim3 g2(BT, 2);
    grpo_finalize_logp<<<g2, 32, 0, stream>>>(x, ref_input, W, ref_W, ids,
                                              partials, logps);

    grpo_loss_reduce<<<1, 256, 0, stream>>>(logps, attn, advantages,
                                            (float*)d_out);
}
